// BasicBlock_29265907155011
// MI455X (gfx1250) — matrix operand order repaired and checked by reading, not compiled, not run on hardware
//
#include <hip/hip_runtime.h>

// ---------------------------------------------------------------------------
// MI455X (gfx1250) sparse-conv BasicBlock:
//   h1 = gatherGEMM(x, nbr1, mask1, W1) + b1          (bf16-split WMMA)
//   y1 = relu(bn1(h1))                                (fused into conv2 gather)
//   h2 = gatherGEMM(y1, nbr2, mask2, W2) + b2         (bf16-split WMMA)
//   out = relu(bn2(h2) + x)
// Precision: fp32 emulated as (hi+lo) bf16 -> Ahi*Bhi + Alo*Bhi + Ahi*Blo
// (3x v_wmma_f32_16x16x32_bf16 per logical fp32 GEMM step, error ~2^-16).
// Each wave owns M=32 rows (two 16x16 D tiles) so one B-fragment load feeds
// two A tiles; B-hi fragments are staged in LDS once per block so the 8 waves
// share them through the DS pipe. The gather is branch-free: nbr rows are
// always valid, so we always load and multiply by mask (0/1) afterwards —
// straight-line inner loop lets the scheduler pipeline loads under WMMAs.
// All cross-block reductions use fixed-order partial buffers (deterministic).
// ---------------------------------------------------------------------------

#define NPTS 400000
#define KOFF 27
#define NCH  32
#define BN_EPS 1e-5f

typedef __attribute__((ext_vector_type(16))) __bf16       v16bf;
typedef __attribute__((ext_vector_type(8)))  float        v8f;
typedef __attribute__((ext_vector_type(8)))  unsigned int v8u;

__device__ __forceinline__ v8f wmma_bf16(v16bf a, v16bf b, v8f c) {
  // 8 args: (neg_a, A, neg_b, B, c_mod, C, reuse_a, reuse_b)
  return __builtin_amdgcn_wmma_f32_16x16x32_bf16(false, a, false, b,
                                                 (short)0, c, false, false);
}

// ---------------------------------------------------------------------------
// Pre-transpose W1/W2 (f32 [27][32][32], W[k][c][n]) into per-lane WMMA
// B-fragment layout, split hi/lo bf16.
// dword index r = ((k*2+f)*32 + lane)*8 + v ; lane half = K-half, v = K pair.
//   c0 = (lane>>4)*8 + 2v + (v>=4 ? 8 : 0) ; n = f*16 + (lane&15)
// dword = {bf16(W[k][c0][n]) , bf16(W[k][c0+1][n])}  (low half = even K)
// ---------------------------------------------------------------------------
__global__ void prep_w_frags(const float* __restrict__ W1,
                             const float* __restrict__ W2,
                             unsigned int* __restrict__ hi1,
                             unsigned int* __restrict__ lo1,
                             unsigned int* __restrict__ hi2,
                             unsigned int* __restrict__ lo2) {
  int tid = blockIdx.x * blockDim.x + threadIdx.x;
  const int per = KOFF * 2 * 32 * 8;        // 13824 dwords per weight set
  if (tid >= 2 * per) return;
  int wsel = tid / per;
  int r    = tid % per;
  int v    = r & 7;
  int lane = (r >> 3) & 31;
  int f    = (r >> 8) & 1;
  int k    = r >> 9;
  int c0   = (lane >> 4) * 8 + 2 * v + ((v >> 2) << 3);
  int n    = f * 16 + (lane & 15);
  const float* W = wsel ? W2 : W1;
  float w0 = W[(k * NCH + c0) * NCH + n];
  float w1 = W[(k * NCH + c0 + 1) * NCH + n];
  __bf16 h0 = (__bf16)w0, h1b = (__bf16)w1;
  __bf16 l0 = (__bf16)(w0 - (float)h0);
  __bf16 l1 = (__bf16)(w1 - (float)h1b);
  unsigned int dh = (unsigned int)__builtin_bit_cast(unsigned short, h0) |
                    ((unsigned int)__builtin_bit_cast(unsigned short, h1b) << 16);
  unsigned int dl = (unsigned int)__builtin_bit_cast(unsigned short, l0) |
                    ((unsigned int)__builtin_bit_cast(unsigned short, l1) << 16);
  if (wsel) { hi2[r] = dh; lo2[r] = dl; }
  else      { hi1[r] = dh; lo1[r] = dl; }
}

// ---------------------------------------------------------------------------
// Branch-free gather of one A fragment (16 rows x 32 K) for point p, kernel
// offset k, split into hi/lo bf16. FUSE_BN applies relu(a*v+c) before the
// mask multiply (matching reference order: mask zeroes the gathered value).
// ---------------------------------------------------------------------------
template <bool FUSE_BN>
__device__ __forceinline__ void gather_a(const float* __restrict__ src,
                                         const int* __restrict__ nbr,
                                         const unsigned char* __restrict__ mask,
                                         int p, int k, int hf,
                                         const float (&ac)[16], const float (&cc)[16],
                                         v16bf& ahi, v16bf& alo) {
  const int idx = p * KOFF + k;
  int   nb = nbr[idx];                       // always a valid row index
  float mf = (float)mask[idx];               // 0.0 or 1.0
  const float4* rp = (const float4*)(src + (size_t)nb * NCH);
  float4 q0 = rp[2 * hf + 0];
  float4 q1 = rp[2 * hf + 1];
  float4 q2 = rp[4 + 2 * hf + 0];
  float4 q3 = rp[4 + 2 * hf + 1];
  float vals[16];
  vals[0]=q0.x;  vals[1]=q0.y;  vals[2]=q0.z;   vals[3]=q0.w;
  vals[4]=q1.x;  vals[5]=q1.y;  vals[6]=q1.z;   vals[7]=q1.w;
  vals[8]=q2.x;  vals[9]=q2.y;  vals[10]=q2.z;  vals[11]=q2.w;
  vals[12]=q3.x; vals[13]=q3.y; vals[14]=q3.z;  vals[15]=q3.w;
#pragma unroll
  for (int i = 0; i < 16; ++i) {
    if (FUSE_BN) vals[i] = fmaxf(fmaf(ac[i], vals[i], cc[i]), 0.f);
    vals[i] *= mf;
    __bf16 h = (__bf16)vals[i];
    ahi[i] = h;
    alo[i] = (__bf16)(vals[i] - (float)h);
  }
}

// ---------------------------------------------------------------------------
// Gather-GEMM-scatter conv. One wave = 32 points (two 16x16 D tiles sharing
// one B-fragment fetch); block = 8 waves = 256 points. B-hi staged in LDS.
// Emits per-block per-channel sum/sumsq partials for this layer's BN.
// ---------------------------------------------------------------------------
template <bool FUSE_BN>
__global__ void __launch_bounds__(256)
conv_kernel(const float* __restrict__ src,
            const int* __restrict__ nbr,
            const unsigned char* __restrict__ mask,   // jax bool = 1 byte
            const unsigned int* __restrict__ fragHi,
            const unsigned int* __restrict__ fragLo,
            const float* __restrict__ bias,
            const float* __restrict__ bn_coef,        // [64]: a[32], shift[32]
            float* __restrict__ hout,                 // [N,32]
            float* __restrict__ stat_part,            // [64 * nBlocks]
            int nBlocks) {
  __shared__ v8u    sBhi[KOFF * 2 * 32];   // 55,296 B: B-hi frags, linear copy
  __shared__ float4 part[256];

  const int tid  = threadIdx.x;
  const int lane = tid & 31;
  const int wv   = tid >> 5;
  const int hf   = lane >> 4;            // K-half of this lane's A fragment
  const int col  = lane & 15;            // row within tile / N column

  // cooperative stage of B-hi fragments into LDS (same linear layout)
  {
    const uint4* g = (const uint4*)fragHi;
    uint4* s = (uint4*)sBhi;
    for (int i = tid; i < KOFF * 2 * 32 * 2; i += 256) s[i] = g[i];
  }
  __syncthreads();

  const int base    = (blockIdx.x * 8 + wv) * 32;  // 32 rows per wave
  const bool active = base < NPTS;                 // uniform per wave

  float4 my = make_float4(0.f, 0.f, 0.f, 0.f);

  if (active) {
    const int pA = base + col;            // tile A gather point
    const int pB = base + 16 + col;       // tile B gather point
    const int cb0 = 8 * hf, cb1 = 16 + 8 * hf;

    float ac[16], cc[16];
    if (FUSE_BN) {
#pragma unroll
      for (int i = 0; i < 16; ++i) {
        int ch = (i < 8) ? (cb0 + i) : (cb1 + i - 8);
        ac[i] = bn_coef[ch];
        cc[i] = bn_coef[32 + ch];
      }
    } else {
#pragma unroll
      for (int i = 0; i < 16; ++i) { ac[i] = 0.f; cc[i] = 0.f; }
    }

    v8f aA0 = {0.f,0.f,0.f,0.f,0.f,0.f,0.f,0.f};
    v8f aA1 = aA0, aB0 = aA0, aB1 = aA0;

    const v8u* fl = (const v8u*)fragLo;

    for (int k = 0; k < KOFF; ++k) {
      int fbase = (k * 2) * 32 + lane;
      v16bf bh0 = __builtin_bit_cast(v16bf, sBhi[fbase]);
      v16bf bh1 = __builtin_bit_cast(v16bf, sBhi[fbase + 32]);
      v16bf bl0 = __builtin_bit_cast(v16bf, fl[fbase]);
      v16bf bl1 = __builtin_bit_cast(v16bf, fl[fbase + 32]);

      v16bf ahi, alo;
      gather_a<FUSE_BN>(src, nbr, mask, pA, k, hf, ac, cc, ahi, alo);
      aA0 = wmma_bf16(ahi, bh0, aA0);
      aA0 = wmma_bf16(alo, bh0, aA0);
      aA0 = wmma_bf16(ahi, bl0, aA0);
      aA1 = wmma_bf16(ahi, bh1, aA1);
      aA1 = wmma_bf16(alo, bh1, aA1);
      aA1 = wmma_bf16(ahi, bl1, aA1);

      gather_a<FUSE_BN>(src, nbr, mask, pB, k, hf, ac, cc, ahi, alo);
      aB0 = wmma_bf16(ahi, bh0, aB0);
      aB0 = wmma_bf16(alo, bh0, aB0);
      aB0 = wmma_bf16(ahi, bl0, aB0);
      aB1 = wmma_bf16(ahi, bh1, aB1);
      aB1 = wmma_bf16(alo, bh1, aB1);
      aB1 = wmma_bf16(ahi, bl1, aB1);
    }

    // epilogue: + bias, per-channel stats, store h
    float b0 = bias[col], b1 = bias[16 + col];
    float s0 = 0.f, q0 = 0.f, s1 = 0.f, q1 = 0.f;
#pragma unroll
    for (int r = 0; r < 8; ++r) {
      aA0[r] += b0; s0 += aA0[r]; q0 += aA0[r] * aA0[r];
      aB0[r] += b0; s0 += aB0[r]; q0 += aB0[r] * aB0[r];
      aA1[r] += b1; s1 += aA1[r]; q1 += aA1[r] * aA1[r];
      aB1[r] += b1; s1 += aB1[r]; q1 += aB1[r] * aB1[r];
    }
    int mA = base + 8 * hf;               // D layout: M = r + 8*(lane/16)
    int mB = base + 16 + 8 * hf;
#pragma unroll
    for (int r = 0; r < 8; ++r) {
      hout[(size_t)(mA + r) * NCH + col]      = aA0[r];
      hout[(size_t)(mA + r) * NCH + 16 + col] = aA1[r];
      hout[(size_t)(mB + r) * NCH + col]      = aB0[r];
      hout[(size_t)(mB + r) * NCH + 16 + col] = aB1[r];
    }
    my = make_float4(s0, q0, s1, q1);
  }

  // deterministic block reduction of channel stats
  part[tid] = my;
  __syncthreads();
  if (tid < 32) {
    int ch = tid;
    int l = ch & 15, fsel = ch >> 4;
    float S = 0.f, Q = 0.f;
    for (int t = l; t < 256; t += 16) {     // fixed ascending order
      float4 v = part[t];
      S += fsel ? v.z : v.x;
      Q += fsel ? v.w : v.y;
    }
    stat_part[ch * nBlocks + blockIdx.x]        = S;
    stat_part[(32 + ch) * nBlocks + blockIdx.x] = Q;
  }
}

// ---------------------------------------------------------------------------
// Fold per-block partials -> BN scale/shift. Fixed-order (deterministic).
// ---------------------------------------------------------------------------
__global__ void bn_finalize(const float* __restrict__ stat_part, int nBlocks,
                            const float* __restrict__ gamma,
                            const float* __restrict__ beta,
                            float* __restrict__ coef /*[64]*/) {
  __shared__ float red[256];
  __shared__ float tot[64];
  int s = threadIdx.x & 63, sub = threadIdx.x >> 6;   // 4 chunks per stat
  int chunk = (nBlocks + 3) / 4;
  int b0 = sub * chunk;
  int b1 = b0 + chunk; if (b1 > nBlocks) b1 = nBlocks;
  float acc = 0.f;
  for (int b = b0; b < b1; ++b) acc += stat_part[s * nBlocks + b];
  red[threadIdx.x] = acc;
  __syncthreads();
  if (threadIdx.x < 64)
    tot[s] = red[s] + red[64 + s] + red[128 + s] + red[192 + s];
  __syncthreads();
  if (threadIdx.x < 32) {
    int c = threadIdx.x;
    float invN = 1.0f / (float)NPTS;
    float mean = tot[c] * invN;
    float var  = tot[32 + c] * invN - mean * mean;
    float a    = gamma[c] * rsqrtf(var + BN_EPS);
    coef[c]      = a;
    coef[32 + c] = beta[c] - mean * a;
  }
}

// ---------------------------------------------------------------------------
// out = relu(a2*h2 + c2 + x), in-place on d_out (h2 pre-stored there).
// ---------------------------------------------------------------------------
__global__ void final_residual(const float* __restrict__ x,
                               const float* __restrict__ coef,
                               float* __restrict__ out, int total4) {
  int i = blockIdx.x * blockDim.x + threadIdx.x;
  if (i >= total4) return;
  const float4* x4 = (const float4*)x;
  float4* o4 = (float4*)out;
  int c4 = i & 7;                                  // float4 index within row
  float4 a = ((const float4*)coef)[c4];
  float4 c = ((const float4*)coef)[8 + c4];
  float4 h = o4[i];
  float4 r = x4[i];
  float4 o;
  o.x = fmaxf(fmaf(a.x, h.x, c.x) + r.x, 0.f);
  o.y = fmaxf(fmaf(a.y, h.y, c.y) + r.y, 0.f);
  o.z = fmaxf(fmaf(a.z, h.z, c.z) + r.z, 0.f);
  o.w = fmaxf(fmaf(a.w, h.w, c.w) + r.w, 0.f);
  o4[i] = o;
}

// ---------------------------------------------------------------------------
extern "C" void kernel_launch(void* const* d_in, const int* in_sizes, int n_in,
                              void* d_out, int out_size, void* d_ws, size_t ws_size,
                              hipStream_t stream) {
  const float* x      = (const float*)d_in[0];
  const float* W1     = (const float*)d_in[1];
  const float* b1     = (const float*)d_in[2];
  const float* gamma1 = (const float*)d_in[3];
  const float* beta1  = (const float*)d_in[4];
  const float* W2     = (const float*)d_in[5];
  const float* b2     = (const float*)d_in[6];
  const float* gamma2 = (const float*)d_in[7];
  const float* beta2  = (const float*)d_in[8];
  const int* nbr1     = (const int*)d_in[9];
  const unsigned char* mask1 = (const unsigned char*)d_in[10];
  const int* nbr2     = (const int*)d_in[11];
  const unsigned char* mask2 = (const unsigned char*)d_in[12];
  float* out = (float*)d_out;

  const int nWaveUnits = (NPTS + 31) / 32;        // 12500 (32 rows per wave)
  const int nBlocks    = (nWaveUnits + 7) / 8;    // 1563
  const size_t fragBytes = (size_t)KOFF * 2 * 32 * 8 * 4;  // 55296

  // workspace carve-up (all offsets 256B aligned); total ~52 MB
  char* ws = (char*)d_ws;
  float* h1           = (float*)ws;                        ws += (size_t)NPTS * NCH * 4;
  unsigned int* fh1   = (unsigned int*)ws;                 ws += fragBytes;
  unsigned int* fl1   = (unsigned int*)ws;                 ws += fragBytes;
  unsigned int* fh2   = (unsigned int*)ws;                 ws += fragBytes;
  unsigned int* fl2   = (unsigned int*)ws;                 ws += fragBytes;
  float* stat         = (float*)ws;                        ws += (size_t)64 * ((nBlocks + 63) & ~63) * 4;
  float* coef1        = (float*)ws;                        ws += 256;
  float* coef2        = (float*)ws;                        ws += 256;
  (void)ws_size; (void)n_in; (void)in_sizes; (void)out_size;

  // 1. weight fragments (hi/lo bf16, WMMA B layout)
  {
    int nth = 2 * KOFF * 2 * 32 * 8;
    prep_w_frags<<<(nth + 255) / 256, 256, 0, stream>>>(W1, W2, fh1, fl1, fh2, fl2);
  }
  // 2. conv1: x -> h1 (+ stats1)
  conv_kernel<false><<<nBlocks, 256, 0, stream>>>(x, nbr1, mask1, fh1, fl1, b1,
                                                  nullptr, h1, stat, nBlocks);
  // 3. bn1 coefficients
  bn_finalize<<<1, 256, 0, stream>>>(stat, nBlocks, gamma1, beta1, coef1);
  // 4. conv2 with fused relu(bn1): h1 -> h2 (stored in d_out) (+ stats2)
  conv_kernel<true><<<nBlocks, 256, 0, stream>>>(h1, nbr2, mask2, fh2, fl2, b2,
                                                 coef1, out, stat, nBlocks);
  // 5. bn2 coefficients
  bn_finalize<<<1, 256, 0, stream>>>(stat, nBlocks, gamma2, beta2, coef2);
  // 6. out = relu(bn2(h2) + x), in place
  {
    int total4 = NPTS * NCH / 4;
    final_residual<<<(total4 + 255) / 256, 256, 0, stream>>>(x, coef2, out, total4);
  }
}
